// DGCNN_51067161149957
// MI455X (gfx1250) — compile-verified, matrix operand order checked by reading, not run on hardware
//
#include <hip/hip_runtime.h>
#include <hip/hip_bf16.h>

typedef __attribute__((ext_vector_type(16))) __bf16 v16bf;
typedef __attribute__((ext_vector_type(4)))  __bf16 v4bf;
typedef __attribute__((ext_vector_type(8)))  float  v8f;

#define HID   64
#define WAVES 4
#define TILE_E (WAVES * 16)

// K-offset of element e (0..15) of a 16-bit WMMA A/B fragment for this lane,
// per CDNA5 ISA 7.12.2 (16-bit 16x32 matrix): VGPR v holds 2 halves;
// v<4 -> K base 0, v>=4 -> K base 16; lanes 16..31 add 8.
__device__ __forceinline__ int kmap16(int e, int lane) {
    int v = e >> 1, h = e & 1;
    return ((v & 4) ? 16 : 0) + ((lane & 16) ? 8 : 0) + ((v & 3) << 1) + h;
}

// Inverse map for a group of 4 consecutive K offsets (koff multiple of 4,
// within one 32-wide K-step): which lane group and element base holds them.
//   lane group = (koff>>3)&1 ;  element base e0 = 2*( (koff>>4)*4 + ((koff>>1)&3) )
__device__ __forceinline__ void kinv4(int koff, int& lgrp, int& e0) {
    lgrp = (koff >> 3) & 1;
    int v = ((koff >> 4) << 2) | ((koff >> 1) & 3);
    e0 = v << 1;
}

__device__ __forceinline__ __bf16 bf(float f) { return (__bf16)f; }

// ---------------------------------------------------------------------------
// Pack a [K,N] f32 weight matrix into WMMA B-fragment-major bf16:
// out[((nt*(K/32)+ks)*32 + lane)*16 + e] = bf16( W[(ks*32+kmap(e,lane))*N + nt*16+(lane&15)] )
// ---------------------------------------------------------------------------
__global__ __launch_bounds__(256) void pack_wfrag_kernel(
    const float* __restrict__ W, __bf16* __restrict__ out, int K, int N)
{
    int t = blockIdx.x * blockDim.x + threadIdx.x;
    int KS = K >> 5, NT = N >> 4;
    if (t >= NT * KS * 32) return;
    int lane = t & 31, f = t >> 5;
    int ks = f % KS, nt = f / KS;
    int col = nt * 16 + (lane & 15);
    __bf16* o = out + (size_t)t * 16;
    #pragma unroll
    for (int e = 0; e < 16; ++e) {
        int k = ks * 32 + kmap16(e, lane);
        o[e] = bf(W[(size_t)k * N + col]);
    }
}

// ---------------------------------------------------------------------------
// h[n,64] = concat(x,pos)[n,6] @ node_W[6,64] + node_b
// ---------------------------------------------------------------------------
__global__ __launch_bounds__(256) void node_encoder_kernel(
    const float* __restrict__ x, const float* __restrict__ pos,
    const float* __restrict__ nW, const float* __restrict__ nb,
    float* __restrict__ h, int nN)
{
    int node = blockIdx.x * blockDim.x + threadIdx.x;
    if (node >= nN) return;
    float f[6] = { x[node*3+0], x[node*3+1], x[node*3+2],
                   pos[node*3+0], pos[node*3+1], pos[node*3+2] };
    #pragma unroll 4
    for (int c = 0; c < HID; ++c) {
        float acc = nb[c];
        #pragma unroll
        for (int j = 0; j < 6; ++j) acc += f[j] * nW[j*HID + c];
        h[(size_t)node*HID + c] = acc;
    }
}

// ---------------------------------------------------------------------------
// cnt[dst] += 1 per edge
// ---------------------------------------------------------------------------
__global__ __launch_bounds__(256) void count_kernel(
    const int* __restrict__ edge_index, float* __restrict__ cnt, int nE)
{
    int e = blockIdx.x * blockDim.x + threadIdx.x;
    if (e < nE) atomicAdd(&cnt[edge_index[nE + e]], 1.0f);
}

// ---------------------------------------------------------------------------
// Fused edge MLP (bf16 WMMA): m = concat(xi, xj-xi, ea) -> GEMM1(192x128)
// -> LN -> ReLU -> GEMM2(128x64) -> LN -> ReLU -> atomic scatter into agg[dst].
// One wave per 16-edge tile; A data staged in LDS in fragment-major order so
// each WMMA operand is a single contiguous 32B load per lane.
// ---------------------------------------------------------------------------
__global__ __launch_bounds__(WAVES * 32) void edge_mlp_kernel(
    const float* __restrict__ h,
    const int*   __restrict__ edge_index,
    const float* __restrict__ edge_attr,
    const float* __restrict__ edge_W, const float* __restrict__ edge_b,
    const __bf16* __restrict__ wf1,   // [8*6*32*16] fragment-major bf16 W1
    const float* __restrict__ b1, const float* __restrict__ g1, const float* __restrict__ be1,
    const __bf16* __restrict__ wf2,   // [4*4*32*16] fragment-major bf16 W2
    const float* __restrict__ b2, const float* __restrict__ g2, const float* __restrict__ be2,
    float* __restrict__ agg, int nE)
{
    // fragment-major A staging: [wave][kstep][lane][elem]
    __shared__ __align__(32) __bf16 sAfrag[WAVES][6][32][16];
    __shared__ float sM1[WAVES][16][128];            // GEMM1 out row-major (f32)
    __shared__ float sMu[WAVES][16], sRs[WAVES][16];
    __shared__ int   sSrc[WAVES][16], sDst[WAVES][16];

    const int lane = threadIdx.x & 31;
    const int w    = threadIdx.x >> 5;
    const int tile = (blockIdx.x * WAVES + w) * 16;
    const v16bf* wfrag1 = (const v16bf*)wf1;
    const v16bf* wfrag2 = (const v16bf*)wf2;

    if (lane < 16) {
        int e = min(tile + lane, nE - 1);
        sSrc[w][lane] = edge_index[e];
        sDst[w][lane] = edge_index[nE + e];
    }
    __syncthreads();

    // --- stage A1 = [xi | xj-xi | ea] as bf16, fragment-major (float4 granules)
    for (int idx4 = lane; idx4 < 16 * 48; idx4 += 32) {
        int r = idx4 / 48, q = idx4 - r * 48;
        int c0 = q * 4;
        float4 v4;
        if (c0 < 64) {
            v4 = *(const float4*)(h + (size_t)sDst[w][r] * HID + c0);
        } else if (c0 < 128) {
            int cc = c0 - 64;
            float4 a = *(const float4*)(h + (size_t)sSrc[w][r] * HID + cc);
            float4 b = *(const float4*)(h + (size_t)sDst[w][r] * HID + cc);
            v4 = make_float4(a.x - b.x, a.y - b.y, a.z - b.z, a.w - b.w);
        } else {
            int cc = c0 - 128;
            int e = min(tile + r, nE - 1);
            const float* a4 = edge_attr + (size_t)e * 4;
            float4 w0 = *(const float4*)(edge_W + cc);
            float4 w1 = *(const float4*)(edge_W + 64 + cc);
            float4 w2 = *(const float4*)(edge_W + 128 + cc);
            float4 w3 = *(const float4*)(edge_W + 192 + cc);
            float4 eb = *(const float4*)(edge_b + cc);
            v4.x = eb.x + a4[0]*w0.x + a4[1]*w1.x + a4[2]*w2.x + a4[3]*w3.x;
            v4.y = eb.y + a4[0]*w0.y + a4[1]*w1.y + a4[2]*w2.y + a4[3]*w3.y;
            v4.z = eb.z + a4[0]*w0.z + a4[1]*w1.z + a4[2]*w2.z + a4[3]*w3.z;
            v4.w = eb.w + a4[0]*w0.w + a4[1]*w1.w + a4[2]*w2.w + a4[3]*w3.w;
        }
        int ks = c0 >> 5, lgrp, e0;
        kinv4(c0 & 31, lgrp, e0);
        *(v4bf*)&sAfrag[w][ks][r + lgrp*16][e0] =
            (v4bf){ bf(v4.x), bf(v4.y), bf(v4.z), bf(v4.w) };
    }
    __syncthreads();

    // --- GEMM1: [16,192] x [192,128]
    v16bf afr[6];
    #pragma unroll
    for (int ks = 0; ks < 6; ++ks)
        afr[ks] = *(const v16bf*)sAfrag[w][ks][lane];

    const int mrow  = lane & 15;
    const int rbase = (lane & 16) ? 8 : 0;
    for (int nt = 0; nt < 8; ++nt) {
        v8f acc = {};
        #pragma unroll
        for (int ks = 0; ks < 6; ++ks) {
            v16bf bfr = wfrag1[(nt * 6 + ks) * 32 + lane];
            acc = __builtin_amdgcn_wmma_f32_16x16x32_bf16(
                      false, afr[ks], false, bfr, (short)0, acc, false, false);
        }
        int col = nt * 16 + mrow;
        float bias = b1[col];
        #pragma unroll
        for (int j = 0; j < 8; ++j)
            sM1[w][rbase + j][col] = acc[j] + bias;
    }
    __syncthreads();

    // --- LN1 row stats over 128
    if (lane < 16) {
        float s = 0.f, s2 = 0.f;
        #pragma unroll 4
        for (int c = 0; c < 128; ++c) { float v = sM1[w][lane][c]; s += v; s2 += v*v; }
        float mu = s * (1.f/128.f);
        float var = s2 * (1.f/128.f) - mu*mu;
        sMu[w][lane] = mu;
        sRs[w][lane] = rsqrtf(var + 1e-5f);
    }
    __syncthreads();

    // --- stage A2 = relu(LN(m1)) bf16 fragment-major (reuse sAfrag, 4 ksteps)
    for (int idx4 = lane; idx4 < 16 * 32; idx4 += 32) {
        int r = idx4 >> 5, q = idx4 & 31;
        int c0 = q * 4;
        float mu = sMu[w][r], rs = sRs[w][r];
        float4 m  = *(const float4*)&sM1[w][r][c0];
        float4 gg = *(const float4*)(g1 + c0);
        float4 bb = *(const float4*)(be1 + c0);
        float4 v4;
        v4.x = fmaxf((m.x - mu) * rs * gg.x + bb.x, 0.f);
        v4.y = fmaxf((m.y - mu) * rs * gg.y + bb.y, 0.f);
        v4.z = fmaxf((m.z - mu) * rs * gg.z + bb.z, 0.f);
        v4.w = fmaxf((m.w - mu) * rs * gg.w + bb.w, 0.f);
        int ks = c0 >> 5, lgrp, e0;
        kinv4(c0 & 31, lgrp, e0);
        *(v4bf*)&sAfrag[w][ks][r + lgrp*16][e0] =
            (v4bf){ bf(v4.x), bf(v4.y), bf(v4.z), bf(v4.w) };
    }
    __syncthreads();

    // --- GEMM2: [16,128] x [128,64]; out row-major into sM1 region (stride 64)
    v16bf a2[4];
    #pragma unroll
    for (int ks = 0; ks < 4; ++ks)
        a2[ks] = *(const v16bf*)sAfrag[w][ks][lane];

    float* sM2 = &sM1[w][0][0];
    for (int nt = 0; nt < 4; ++nt) {
        v8f acc = {};
        #pragma unroll
        for (int ks = 0; ks < 4; ++ks) {
            v16bf bfr = wfrag2[(nt * 4 + ks) * 32 + lane];
            acc = __builtin_amdgcn_wmma_f32_16x16x32_bf16(
                      false, a2[ks], false, bfr, (short)0, acc, false, false);
        }
        int col = nt * 16 + mrow;
        float bias = b2[col];
        #pragma unroll
        for (int j = 0; j < 8; ++j)
            sM2[(rbase + j) * 64 + col] = acc[j] + bias;
    }
    __syncthreads();

    // --- LN2 row stats over 64
    if (lane < 16) {
        float s = 0.f, s2 = 0.f;
        #pragma unroll 4
        for (int c = 0; c < 64; ++c) { float v = sM2[lane*64 + c]; s += v; s2 += v*v; }
        float mu = s * (1.f/64.f);
        float var = s2 * (1.f/64.f) - mu*mu;
        sMu[w][lane] = mu;
        sRs[w][lane] = rsqrtf(var + 1e-5f);
    }
    __syncthreads();

    // --- LN + affine + ReLU + atomic scatter-add into agg[dst]
    for (int idx = lane; idx < 16 * 64; idx += 32) {
        int r = idx >> 6, c = idx & 63;
        if (tile + r < nE) {
            float v = (sM2[r*64 + c] - sMu[w][r]) * sRs[w][r] * g2[c] + be2[c];
            atomicAdd(&agg[(size_t)sDst[w][r]*HID + c], fmaxf(v, 0.f));
        }
    }
}

// ---------------------------------------------------------------------------
// o = relu(LN((agg/cnt) @ post_W + post_b)); h += o
// ---------------------------------------------------------------------------
__global__ __launch_bounds__(128) void node_update_kernel(
    const float* __restrict__ agg, const float* __restrict__ cnt,
    const float* __restrict__ pW, const float* __restrict__ pb,
    const float* __restrict__ og, const float* __restrict__ obe,
    float* __restrict__ h, int nN)
{
    int node = blockIdx.x * blockDim.x + threadIdx.x;
    if (node >= nN) return;
    float a[HID];
    float invc = 1.f / fmaxf(cnt[node], 1.f);
    #pragma unroll 8
    for (int k = 0; k < HID; ++k) a[k] = agg[(size_t)node*HID + k] * invc;

    float o[HID];
    float s = 0.f, s2 = 0.f;
    for (int n = 0; n < HID; ++n) {
        float acc = pb[n];
        #pragma unroll 8
        for (int k = 0; k < HID; ++k) acc += a[k] * pW[k*HID + n];
        o[n] = acc; s += acc; s2 += acc*acc;
    }
    float mu = s * (1.f/64.f);
    float rs = rsqrtf(s2 * (1.f/64.f) - mu*mu + 1e-5f);
    #pragma unroll 8
    for (int n = 0; n < HID; ++n) {
        float v = (o[n] - mu) * rs * og[n] + obe[n];
        h[(size_t)node*HID + n] += fmaxf(v, 0.f);
    }
}

// ---------------------------------------------------------------------------
extern "C" void kernel_launch(void* const* d_in, const int* in_sizes, int n_in,
                              void* d_out, int out_size, void* d_ws, size_t ws_size,
                              hipStream_t stream)
{
    const float* x         = (const float*)d_in[0];
    const float* pos       = (const float*)d_in[1];
    const float* edge_attr = (const float*)d_in[2];
    const int*   edge_index= (const int*)  d_in[3];
    // d_in[4] = batch (unused; single graph)
    const float* node_W  = (const float*)d_in[5];
    const float* node_b  = (const float*)d_in[6];
    const float* edge_W  = (const float*)d_in[7];
    const float* edge_b  = (const float*)d_in[8];
    const float* mlp_W1  = (const float*)d_in[9];
    const float* mlp_b1  = (const float*)d_in[10];
    const float* mlp_g1  = (const float*)d_in[11];
    const float* mlp_be1 = (const float*)d_in[12];
    const float* mlp_W2  = (const float*)d_in[13];
    const float* mlp_b2  = (const float*)d_in[14];
    const float* mlp_g2  = (const float*)d_in[15];
    const float* mlp_be2 = (const float*)d_in[16];
    const float* post_W  = (const float*)d_in[17];
    const float* post_b  = (const float*)d_in[18];
    const float* out_g   = (const float*)d_in[19];
    const float* out_be  = (const float*)d_in[20];

    const int N = in_sizes[0] / 3;
    const int E = in_sizes[2] / 4;

    float* hbuf = (float*)d_out;                         // h lives in d_out
    char*  ws   = (char*)d_ws;
    float* agg  = (float*)ws;                            // [N,64] f32
    float* cnt  = (float*)(ws + (size_t)N * HID * 4);    // [N] f32
    __bf16* wf1 = (__bf16*)(ws + (size_t)N * HID * 4 + (size_t)N * 4);  // 2x 24576 bf16
    __bf16* wf2 = wf1 + 2 * 192 * 128;                                   // 2x 8192 bf16

    hipMemsetAsync(cnt, 0, (size_t)N * sizeof(float), stream);
    for (int l = 0; l < 2; ++l) {
        pack_wfrag_kernel<<<(8*6*32 + 255) / 256, 256, 0, stream>>>(
            mlp_W1 + (size_t)l * 192 * 128, wf1 + (size_t)l * 192 * 128, 192, 128);
        pack_wfrag_kernel<<<(4*4*32 + 255) / 256, 256, 0, stream>>>(
            mlp_W2 + (size_t)l * 128 * 64,  wf2 + (size_t)l * 128 * 64,  128, 64);
    }
    node_encoder_kernel<<<(N + 255) / 256, 256, 0, stream>>>(x, pos, node_W, node_b, hbuf, N);
    count_kernel<<<(E + 255) / 256, 256, 0, stream>>>(edge_index, cnt, E);

    for (int l = 0; l < 2; ++l) {
        hipMemsetAsync(agg, 0, (size_t)N * HID * sizeof(float), stream);
        edge_mlp_kernel<<<(E + TILE_E - 1) / TILE_E, WAVES * 32, 0, stream>>>(
            hbuf, edge_index, edge_attr, edge_W, edge_b,
            wf1 + (size_t)l * 192 * 128,
            mlp_b1 + l * 128, mlp_g1 + l * 128, mlp_be1 + l * 128,
            wf2 + (size_t)l * 128 * 64,
            mlp_b2 + l * 64,  mlp_g2 + l * 64,  mlp_be2 + l * 64,
            agg, E);
        node_update_kernel<<<(N + 127) / 128, 128, 0, stream>>>(
            agg, cnt,
            post_W + (size_t)l * 64 * 64, post_b + l * 64,
            out_g + l * 64, out_be + l * 64, hbuf, N);
    }
}
